// GPT_10926396801388
// MI455X (gfx1250) — matrix operand order repaired and checked by reading, not compiled, not run on hardware
//
#include <hip/hip_runtime.h>
#include <hip/hip_bf16.h>
#include <math.h>

// ---------------------------------------------------------------------------
// GPT forward for gfx1250 (MI455X). All GEMMs + attention use
// v_wmma_f32_16x16x32_f16 (f16 multiply, f32 accumulate).
// ---------------------------------------------------------------------------

typedef _Float16 v16h __attribute__((ext_vector_type(16)));
typedef _Float16 h8   __attribute__((ext_vector_type(8)));
typedef float    v8f  __attribute__((ext_vector_type(8)));
typedef float    f4   __attribute__((ext_vector_type(4)));

#define WMMA16(a, b, c) \
  __builtin_amdgcn_wmma_f32_16x16x32_f16(false, (a), false, (b), (short)0, (c), false, false)

namespace gptk {
constexpr int Bc = 4, T = 1024, BT = 4096, E = 512, H = 8, HS = 64, FF = 2048, NL = 4, V = 50257;
}

// ---------------------------------------------------------------------------
// Embedding: x[b,t,:] = tok_emb[idx[b,t],:] + pos_emb[t,:]
// ---------------------------------------------------------------------------
__global__ __launch_bounds__(256) void embed_kernel(const int* __restrict__ idx,
                                                    const float* __restrict__ tok,
                                                    const float* __restrict__ pos,
                                                    float* __restrict__ x) {
  using namespace gptk;
  int i = blockIdx.x * 256 + threadIdx.x;          // i < BT*E
  int bt = i >> 9;                                  // E == 512
  int e  = i & 511;
  int t  = bt & (T - 1);
  x[i] = tok[(size_t)idx[bt] * E + e] + pos[(size_t)t * E + e];
}

// ---------------------------------------------------------------------------
// LayerNorm over E=512, one block (256 thr) per row; emits f16 activations.
// ---------------------------------------------------------------------------
__global__ __launch_bounds__(256) void ln_kernel(const float* __restrict__ x,
                                                 const float* __restrict__ g,
                                                 const float* __restrict__ bta,
                                                 _Float16* __restrict__ out) {
  using namespace gptk;
  int row = blockIdx.x, tid = threadIdx.x;
  __shared__ float red[256];
  const float* xr = x + (size_t)row * E;
  float v0 = xr[tid], v1 = xr[tid + 256];
  red[tid] = v0 + v1;
  __syncthreads();
  for (int s = 128; s > 0; s >>= 1) { if (tid < s) red[tid] += red[tid + s]; __syncthreads(); }
  float mu = red[0] * (1.0f / E);
  __syncthreads();
  float d0 = v0 - mu, d1 = v1 - mu;
  red[tid] = d0 * d0 + d1 * d1;
  __syncthreads();
  for (int s = 128; s > 0; s >>= 1) { if (tid < s) red[tid] += red[tid + s]; __syncthreads(); }
  float rstd = rsqrtf(red[0] * (1.0f / E) + 1e-5f);
  out[(size_t)row * E + tid]       = (_Float16)(d0 * rstd * g[tid] + bta[tid]);
  out[(size_t)row * E + tid + 256] = (_Float16)(d1 * rstd * g[tid + 256] + bta[tid + 256]);
}

// ---------------------------------------------------------------------------
// Fragment load helpers.
// A fragment (16-bit A 16x32 layout): lane-low = row, lane-half = K-subrange.
// ---------------------------------------------------------------------------
__device__ __forceinline__ v16h load_a_frag(const _Float16* arow) {
  h8 alo = *(const h8*)(arow);        // K = 8*lh + 0..7   (caller bakes lh in)
  h8 ahi = *(const h8*)(arow + 16);   // K = 16 + 8*lh + 0..7
  v16h a;
#pragma unroll
  for (int i = 0; i < 8; ++i) { a[i] = alo[i]; a[8 + i] = ahi[i]; }
  return a;
}

// B fragment fast path: one column, K rows 8*lh+0..7 and 16+8*lh+0..7 (stride ld).
__device__ __forceinline__ v16h load_b_frag(const float* p, size_t ld) {
  v16h b;
#pragma unroll
  for (int i = 0; i < 8; ++i) {
    b[i]     = (_Float16)p[(size_t)i * ld];
    b[8 + i] = (_Float16)p[(size_t)(16 + i) * ld];
  }
  return b;
}

// ---------------------------------------------------------------------------
// Generic WMMA GEMM:  out[M,N] = act( Af16[M,K] @ W + bias [+resid] )
// Weight element (k,n) lives at W[k*wsk + (n/nblk)*wsb + n%nblk]:
//   row-major [K,N]      : wsk=N,  wsb=0,      nblk=N
//   QKV [H,E,HS] layout  : wsk=HS, wsb=E*HS,   nblk=HS
// Block: 256 thr = 8 waves (2x4). Wave tile 32x32 (2x2 WMMA accumulators),
// block tile 64(M) x 128(N). K stepped by 32; the K-loop is unrolled by 2
// with ping-pong fragment buffers so the software pipeline carries no
// register-rotation movs. Requires K % 64 == 0 (true for K = 512, 2048).
// ---------------------------------------------------------------------------
__global__ __launch_bounds__(256) void gemm_kernel(const _Float16* __restrict__ A,
                                                   const float* __restrict__ W,
                                                   const float* __restrict__ bias,
                                                   const float* __restrict__ resid,
                                                   float* __restrict__ outF,
                                                   _Float16* __restrict__ outH,
                                                   int M, int N, int K,
                                                   int wsk, int wsb, int nblk, int relu) {
  int lane = threadIdx.x & 31;
  int wave = threadIdx.x >> 5;
  int ll = lane & 15, lh = lane >> 4;           // lane low / lane half
  int wr = wave >> 2, wc = wave & 3;
  int mbase = blockIdx.y * 64 + wr * 32;
  int nbase = blockIdx.x * 128 + wc * 32;
  if (nbase >= N) return;                        // whole wave out of range (no barriers here)

  int krow_lane = 8 * lh;                        // B fragment K-row base (table order)
  int c0 = nbase + ll, c1 = nbase + 16 + ll;

  // A row pointers for the two M-fragments (16B aligned: K multiple of 32).
  const _Float16* pa0 = A + (size_t)(mbase + ll) * K + 8 * lh;
  const _Float16* pa1 = pa0 + (size_t)16 * K;

  v8f acc00 = {}, acc01 = {}, acc10 = {}, acc11 = {};
  int steps = K / 32;                            // even for all our shapes

  if (nbase + 32 <= N) {
    // ---------------- fast path: contiguous vectorized B loads ------------
    const float* pb0 = W + (size_t)krow_lane * wsk + (size_t)(c0 / nblk) * wsb + (c0 % nblk);
    const float* pb1 = W + (size_t)krow_lane * wsk + (size_t)(c1 / nblk) * wsb + (c1 % nblk);
    size_t bstep = (size_t)32 * wsk;

    // Ping-pong buffers: set X computes while set Y loads, and vice versa.
    v16h a0x = load_a_frag(pa0), a1x = load_a_frag(pa1);
    v16h b0x = load_b_frag(pb0, wsk), b1x = load_b_frag(pb1, wsk);
    pa0 += 32; pa1 += 32; pb0 += bstep; pb1 += bstep;

    for (int s = 0; s < steps; s += 2) {
      // Load step s+1 (always valid: steps is even).
      v16h a0y = load_a_frag(pa0), a1y = load_a_frag(pa1);
      v16h b0y = load_b_frag(pb0, wsk), b1y = load_b_frag(pb1, wsk);
      __builtin_prefetch(pb0 + bstep, 0, 3);
      pa0 += 32; pa1 += 32; pb0 += bstep; pb1 += bstep;

      acc00 = WMMA16(a0x, b0x, acc00);
      acc01 = WMMA16(a0x, b1x, acc01);
      acc10 = WMMA16(a1x, b0x, acc10);
      acc11 = WMMA16(a1x, b1x, acc11);

      if (s + 2 < steps) {                       // load step s+2 into set X
        a0x = load_a_frag(pa0); a1x = load_a_frag(pa1);
        b0x = load_b_frag(pb0, wsk); b1x = load_b_frag(pb1, wsk);
        pa0 += 32; pa1 += 32; pb0 += bstep; pb1 += bstep;
      }
      acc00 = WMMA16(a0y, b0y, acc00);
      acc01 = WMMA16(a0y, b1y, acc01);
      acc10 = WMMA16(a1y, b0y, acc10);
      acc11 = WMMA16(a1y, b1y, acc11);
    }
  } else {
    // ---------------- tail path (rare): clamped-address scalar loads ------
    // Clamp keeps every load in-bounds; out-of-range columns produce garbage
    // that is never stored (output col n depends only on B col n).
    for (int s = 0; s < steps; ++s) {
      int krow = s * 32 + krow_lane;
      v16h a0 = load_a_frag(pa0), a1 = load_a_frag(pa1);
      pa0 += 32; pa1 += 32;
      v16h b0, b1;
#pragma unroll
      for (int i = 0; i < 16; ++i) {
        int col0 = c0; col0 = col0 < N ? col0 : N - 1;
        int col1 = c1; col1 = col1 < N ? col1 : N - 1;
        int kr = krow + (i < 8 ? i : i + 8);
        b0[i] = (_Float16)W[(size_t)kr * wsk + (size_t)(col0 / nblk) * wsb + (col0 % nblk)];
        b1[i] = (_Float16)W[(size_t)kr * wsk + (size_t)(col1 / nblk) * wsb + (col1 % nblk)];
      }
      acc00 = WMMA16(a0, b0, acc00);
      acc01 = WMMA16(a0, b1, acc01);
      acc10 = WMMA16(a1, b0, acc10);
      acc11 = WMMA16(a1, b1, acc11);
    }
  }

  // ---- Epilogue (C layout: row = r + 8*lh, col = ll).
#pragma unroll
  for (int ni = 0; ni < 2; ++ni) {
    int n = nbase + 16 * ni + ll;
    if (n >= N) continue;
    float bv = bias ? bias[n] : 0.0f;
#pragma unroll
    for (int mi = 0; mi < 2; ++mi) {
      const v8f& acc = (mi == 0) ? (ni == 0 ? acc00 : acc01)
                                 : (ni == 0 ? acc10 : acc11);
#pragma unroll
      for (int r = 0; r < 8; ++r) {
        int mr = mbase + 16 * mi + r + 8 * lh;
        float v = acc[r] + bv;
        if (resid) v += resid[(size_t)mr * N + n];
        if (relu) v = fmaxf(v, 0.0f);
        if (outF) outF[(size_t)mr * N + n] = v;
        if (outH) outH[(size_t)mr * N + n] = (_Float16)v;
      }
    }
  }
}

// ---------------------------------------------------------------------------
// Flash attention: 1 wave per (b, h, 16 q-rows). K/V tiles staged in LDS with
// vectorized loads; online softmax via LDS; S and P*V on WMMA.
// q,k,v are f16 in [BT, E] layout (col = h*HS + d).  Output o f16 same layout.
// ---------------------------------------------------------------------------
__global__ __launch_bounds__(32) void attn_kernel(const _Float16* __restrict__ q,
                                                  const _Float16* __restrict__ k,
                                                  const _Float16* __restrict__ v,
                                                  const int* __restrict__ amask,
                                                  _Float16* __restrict__ o) {
  using namespace gptk;
  int tq = blockIdx.x * 16;
  int h  = blockIdx.y;
  int b  = blockIdx.z;
  int lane = threadIdx.x;
  int ll = lane & 15, lh = lane >> 4;

  __shared__ _Float16 sK[32][64];     // 32 keys x HS
  __shared__ _Float16 sV[32][64];
  __shared__ float    sS[16][32];
  __shared__ _Float16 sP[16][32];
  __shared__ float    sMax[16], sSum[16], sCorr[16];

  size_t base = ((size_t)b * T) * E + (size_t)h * HS;   // + t*E + d

  // Q fragments (A-matrix 16x32 f16), two K-halves covering d = 0..63.
  v16h aq[2];
  {
    const _Float16* qr = q + base + (size_t)(tq + ll) * E;
#pragma unroll
    for (int j = 0; j < 2; ++j) {
      h8 lo = *(const h8*)(qr + 32 * j + 8 * lh);
      h8 hi = *(const h8*)(qr + 32 * j + 16 + 8 * lh);
#pragma unroll
      for (int i = 0; i < 8; ++i) { aq[j][i] = lo[i]; aq[j][8 + i] = hi[i]; }
    }
  }

  if (lane < 16) { sMax[lane] = -INFINITY; sSum[lane] = 0.0f; }
  __syncthreads();

  v8f o0 = {}, o1 = {}, o2 = {}, o3 = {};
  const float scale = 0.125f;   // HS^-0.5 = 1/8
  int krow = ll + 16 * lh;      // B-fragment K-row for this lane (0..31)

  for (int s0 = 0; s0 < tq + 16; s0 += 32) {
    // ---- Stage K/V tiles: lane loads one key's 64 halves (8x b128 each).
    {
      const _Float16* kr = k + base + (size_t)(s0 + lane) * E;
      const _Float16* vr = v + base + (size_t)(s0 + lane) * E;
#pragma unroll
      for (int j = 0; j < 8; ++j) {
        *(h8*)&sK[lane][8 * j] = *(const h8*)(kr + 8 * j);
        *(h8*)&sV[lane][8 * j] = *(const h8*)(vr + 8 * j);
      }
    }
    __syncthreads();

    // ---- S = Q @ K^T for 32 keys (two 16-key chunks, d-contraction = 2x32).
#pragma unroll
    for (int c = 0; c < 2; ++c) {
      v8f sacc = {};
#pragma unroll
      for (int j = 0; j < 2; ++j) {
        int d = 32 * j + 8 * lh;               // B lane = key column; elements = head dim (table order)
        v16h bk;
#pragma unroll
        for (int i = 0; i < 16; ++i) bk[i] = sK[16 * c + ll][d + (i < 8 ? i : i + 8)];
        sacc = WMMA16(aq[j], bk, sacc);
      }
#pragma unroll
      for (int r = 0; r < 8; ++r) {
        int mr = r + 8 * lh;
        int tt = tq + mr, ss = s0 + 16 * c + ll;
        bool valid = (ss <= tt) && (amask[b * T + ss] != 0);
        sS[mr][16 * c + ll] = valid ? sacc[r] * scale : -INFINITY;
      }
    }
    __syncthreads();

    // ---- Online softmax update (one thread per q-row).
    if (lane < 16) {
      float tm = -INFINITY;
#pragma unroll
      for (int j = 0; j < 32; ++j) tm = fmaxf(tm, sS[lane][j]);
      float om = sMax[lane];
      float nm = fmaxf(om, tm);
      float corr = __expf(om - nm);
      float psum = 0.0f;
#pragma unroll
      for (int j = 0; j < 32; ++j) {
        float p = __expf(sS[lane][j] - nm);
        sP[lane][j] = (_Float16)p;
        psum += p;
      }
      sMax[lane] = nm;
      sSum[lane] = sSum[lane] * corr + psum;
      sCorr[lane] = corr;
    }
    __syncthreads();

    // ---- Rescale accumulators, then O += P @ V.
#pragma unroll
    for (int r = 0; r < 8; ++r) {
      float cr = sCorr[r + 8 * lh];
      o0[r] *= cr; o1[r] *= cr; o2[r] *= cr; o3[r] *= cr;
    }
    v16h ap;
#pragma unroll
    for (int i = 0; i < 8; ++i) {
      ap[i]     = sP[ll][8 * lh + i];
      ap[8 + i] = sP[ll][16 + 8 * lh + i];
    }
#pragma unroll
    for (int nc = 0; nc < 4; ++nc) {
      v16h bvf;
#pragma unroll
      for (int i = 0; i < 8; ++i) {
        bvf[i]     = sV[8 * lh + i][nc * 16 + ll];
        bvf[8 + i] = sV[16 + 8 * lh + i][nc * 16 + ll];
      }
      v8f* oc = (nc == 0) ? &o0 : (nc == 1) ? &o1 : (nc == 2) ? &o2 : &o3;
      *oc = WMMA16(ap, bvf, *oc);
    }
    __syncthreads();
  }

  // ---- Normalize and store (C layout).
#pragma unroll
  for (int r = 0; r < 8; ++r) {
    int mr = r + 8 * lh;
    float inv = 1.0f / sSum[mr];
    size_t ob = base + (size_t)(tq + mr) * E + ll;
    o[ob + 0]  = (_Float16)(o0[r] * inv);
    o[ob + 16] = (_Float16)(o1[r] * inv);
    o[ob + 32] = (_Float16)(o2[r] * inv);
    o[ob + 48] = (_Float16)(o3[r] * inv);
  }
}

// ---------------------------------------------------------------------------
// Loss: per-row logsumexp over V, mean NLL accumulated atomically.
// ---------------------------------------------------------------------------
__global__ void zero_loss_kernel(float* loss) { if (threadIdx.x == 0) *loss = 0.0f; }

__global__ __launch_bounds__(256) void loss_kernel(const float* __restrict__ logits,
                                                   const int* __restrict__ targets,
                                                   float* __restrict__ loss) {
  using namespace gptk;
  int row = blockIdx.x, tid = threadIdx.x;
  const float* lr = logits + (size_t)row * V;
  __shared__ float red[256];
  float m = -INFINITY;
  for (int j = tid; j < V; j += 256) m = fmaxf(m, lr[j]);
  red[tid] = m;
  __syncthreads();
  for (int s = 128; s > 0; s >>= 1) { if (tid < s) red[tid] = fmaxf(red[tid], red[tid + s]); __syncthreads(); }
  m = red[0];
  __syncthreads();
  float sum = 0.0f;
  for (int j = tid; j < V; j += 256) sum += __expf(lr[j] - m);
  red[tid] = sum;
  __syncthreads();
  for (int s = 128; s > 0; s >>= 1) { if (tid < s) red[tid] += red[tid + s]; __syncthreads(); }
  if (tid == 0) {
    float lse = logf(red[0]) + m;
    float li = lse - lr[targets[row]];
    atomicAdd(loss, li * (1.0f / BT));
  }
}

// ---------------------------------------------------------------------------
// Host-side orchestration
// ---------------------------------------------------------------------------
static inline void launch_gemm(hipStream_t s, const _Float16* A, const float* W,
                               const float* bias, const float* resid,
                               float* outF, _Float16* outH,
                               int M, int N, int K, int wsk, int wsb, int nblk, int relu) {
  dim3 grid((N + 127) / 128, M / 64);
  gemm_kernel<<<grid, 256, 0, s>>>(A, W, bias, resid, outF, outH, M, N, K, wsk, wsb, nblk, relu);
}

extern "C" void kernel_launch(void* const* d_in, const int* in_sizes, int n_in,
                              void* d_out, int out_size, void* d_ws, size_t ws_size,
                              hipStream_t stream) {
  using namespace gptk;
  const int*   idx   = (const int*)  d_in[0];
  const int*   tgt   = (const int*)  d_in[1];
  const int*   amask = (const int*)  d_in[2];
  const float* tok   = (const float*)d_in[3];
  const float* pos   = (const float*)d_in[4];
  const float* wq    = (const float*)d_in[5];
  const float* wk    = (const float*)d_in[6];
  const float* wv    = (const float*)d_in[7];
  const float* bq    = (const float*)d_in[8];
  const float* bk    = (const float*)d_in[9];
  const float* bv    = (const float*)d_in[10];
  const float* wo    = (const float*)d_in[11];
  const float* bo    = (const float*)d_in[12];
  const float* ln1g  = (const float*)d_in[13];
  const float* ln1b  = (const float*)d_in[14];
  const float* ln2g  = (const float*)d_in[15];
  const float* ln2b  = (const float*)d_in[16];
  const float* w1    = (const float*)d_in[17];
  const float* b1    = (const float*)d_in[18];
  const float* w2    = (const float*)d_in[19];
  const float* b2    = (const float*)d_in[20];
  const float* lnfg  = (const float*)d_in[21];
  const float* lnfb  = (const float*)d_in[22];
  const float* wf    = (const float*)d_in[23];
  const float* bf    = (const float*)d_in[24];

  // Workspace layout (bytes): f16 activation copies + f32 residual stream.
  char* ws = (char*)d_ws;
  const size_t MB = 1u << 20;
  _Float16* h16 = (_Float16*)(ws + 0 * MB);    // [BT,E]  4 MB  (LN output)
  _Float16* q16 = (_Float16*)(ws + 4 * MB);    // [BT,E]  4 MB
  _Float16* k16 = (_Float16*)(ws + 8 * MB);    // [BT,E]  4 MB
  _Float16* v16 = (_Float16*)(ws + 12 * MB);   // [BT,E]  4 MB
  _Float16* o16 = (_Float16*)(ws + 16 * MB);   // [BT,E]  4 MB  (attention out)
  _Float16* m16 = (_Float16*)(ws + 20 * MB);   // [BT,FF] 16 MB (MLP hidden)
  float*    x   = (float*)   (ws + 36 * MB);   // [BT,E]  8 MB  (residual stream)

  float* logits = (float*)d_out;               // [BT, V]
  float* loss   = logits + (size_t)BT * V;     // scalar

  // 1) Embeddings.
  embed_kernel<<<(BT * E) / 256, 256, 0, stream>>>(idx, tok, pos, x);

  // 2) Transformer layers.
  for (int l = 0; l < NL; ++l) {
    const float* wq_l = wq + (size_t)l * H * E * HS;
    const float* wk_l = wk + (size_t)l * H * E * HS;
    const float* wv_l = wv + (size_t)l * H * E * HS;

    ln_kernel<<<BT, 256, 0, stream>>>(x, ln1g + l * E, ln1b + l * E, h16);

    // QKV projections: weight layout [H,E,HS] -> wsk=HS, wsb=E*HS, nblk=HS.
    launch_gemm(stream, h16, wq_l, bq + l * E, nullptr, nullptr, q16, BT, E, E, HS, E * HS, HS, 0);
    launch_gemm(stream, h16, wk_l, bk + l * E, nullptr, nullptr, k16, BT, E, E, HS, E * HS, HS, 0);
    launch_gemm(stream, h16, wv_l, bv + l * E, nullptr, nullptr, v16, BT, E, E, HS, E * HS, HS, 0);

    // Flash attention.
    dim3 ag(T / 16, H, Bc);
    attn_kernel<<<ag, 32, 0, stream>>>(q16, k16, v16, amask, o16);

    // Output projection with residual: x = x + o @ wo + bo.
    launch_gemm(stream, o16, wo + (size_t)l * E * E, bo + l * E, x, x, nullptr, BT, E, E, E, 0, E, 0);

    // MLP.
    ln_kernel<<<BT, 256, 0, stream>>>(x, ln2g + l * E, ln2b + l * E, h16);
    launch_gemm(stream, h16, w1 + (size_t)l * E * FF, b1 + l * FF, nullptr, nullptr, m16, BT, FF, E, FF, 0, FF, 1);
    launch_gemm(stream, m16, w2 + (size_t)l * FF * E, b2 + l * E, x, x, nullptr, BT, E, FF, E, 0, E, 0);
  }

  // 3) Final LN + logits GEMM straight into d_out.
  ln_kernel<<<BT, 256, 0, stream>>>(x, lnfg, lnfb, h16);
  launch_gemm(stream, h16, wf, bf, nullptr, logits, nullptr, BT, V, E, V, 0, V, 0);

  // 4) Loss.
  zero_loss_kernel<<<1, 32, 0, stream>>>(loss);
  loss_kernel<<<BT, 256, 0, stream>>>(logits, tgt, loss);
}
